// cutoff_deltanet_majority_vote_30545807409261
// MI455X (gfx1250) — matrix operand order repaired and checked by reading, not compiled, not run on hardware
//
#include <hip/hip_runtime.h>
#include <hip/hip_bf16.h>
#include <cstdint>

typedef __attribute__((ext_vector_type(16))) _Float16 v16h;
typedef __attribute__((ext_vector_type(8)))  _Float16 v8h;
typedef __attribute__((ext_vector_type(8)))  float    v8f;

#define T_LEN 1024
#define BATCH 32
#define HID   256

__device__ __forceinline__ float sigmoidf_(float x) { return 1.0f / (1.0f + __expf(-x)); }

// ---------------------------------------------------------------------------
// Pack W[N,K] (f32 row-major, torch Linear layout) into WMMA B-fragment order.
// Fragment (n_tile, kstep): 32 lanes x 16 halves, lane%16 = N column within n_tile,
// half j = K row (lane/16)*8 + (j&7) + (j>>3)*16 within kstep (table order).
// Zero-pads the K tail (only W0 with K=80 needs it).
// ---------------------------------------------------------------------------
__global__ void pack_w_kernel(const float* __restrict__ W, _Float16* __restrict__ P,
                              int N, int K) {
  const int ksteps = (K + 31) >> 5;
  const int total  = (N >> 4) * ksteps * 32;
  const int tid = blockIdx.x * blockDim.x + threadIdx.x;
  if (tid >= total) return;
  const int lane = tid & 31;
  const int frag = tid >> 5;
  const int ks   = frag % ksteps;
  const int nt   = frag / ksteps;
  const int n    = nt * 16 + (lane & 15);
  _Float16* dst = P + (size_t)tid * 16;
#pragma unroll
  for (int j = 0; j < 16; ++j) {
    const int k = ks * 32 + (lane >> 4) * 8 + (j & 7) + (j >> 3) * 16;
    const float v = (k < K) ? W[(size_t)n * K + k] : 0.0f;
    dst[j] = (_Float16)v;
  }
}

// ---------------------------------------------------------------------------
// x[32768,80] f32 -> xp[32768,96] f16 (zero-padded K so no guards in GEMM).
// ---------------------------------------------------------------------------
__global__ void pad_x_kernel(const float* __restrict__ x, _Float16* __restrict__ xp) {
  const int idx = blockIdx.x * blockDim.x + threadIdx.x;   // 32768*96
  if (idx >= BATCH * T_LEN * 96) return;
  const int c   = idx % 96;
  const int row = idx / 96;
  xp[idx] = (c < 80) ? (_Float16)x[(size_t)row * 80 + c] : (_Float16)0.0f;
}

// ---------------------------------------------------------------------------
// GEMM: act( A16[rowmap[m], 0:K] @ W^T + b1 (+ b2) ), K % 32 == 0, A in f16.
// One wave computes a 32x64 tile (2 M-tiles x 4 N-tiles): each B fragment
// feeds two WMMAs (8 WMMAs per k-step vs 8x b128 B loads + 4x b128 A loads).
// A fragment = two contiguous 8-half runs -> 2x global_load_b128, no VALU
// converts in the loop. Output optionally written as f32 and/or f16.
// ---------------------------------------------------------------------------
__global__ void wmma_gemm_kernel(const _Float16* __restrict__ A,
                                 const int* __restrict__ rowmap,
                                 const _Float16* __restrict__ Wp,
                                 const float* __restrict__ b1,
                                 const float* __restrict__ b2,
                                 float* __restrict__ C32,
                                 _Float16* __restrict__ C16,
                                 int M, int N, int K, int lda, int ldc, int relu) {
  const int lane = threadIdx.x & 31;
  const int wave = threadIdx.x >> 5;
  const int NG = N >> 6;                        // groups of 4 n-tiles
  const int g  = blockIdx.x * (blockDim.x >> 5) + wave;
  if (g >= (M >> 5) * NG) return;               // grids sized exactly: never diverges
  const int mt2 = g / NG;                       // 32-row tile pair
  const int ng  = g % NG;
  const int ksteps = K >> 5;
  const int hi = lane >> 4;

  const int row0 = mt2 * 32 + (lane & 15);
  const int row1 = row0 + 16;
  const int ri0  = rowmap ? rowmap[row0] : row0;
  const int ri1  = rowmap ? rowmap[row1] : row1;
  const _Float16* Ar0 = A + (size_t)ri0 * lda + hi * 8;
  const _Float16* Ar1 = A + (size_t)ri1 * lda + hi * 8;

  const size_t bstride = (size_t)ksteps * 512;  // halves per n-tile
  const _Float16* bp0 = Wp + ((size_t)((ng * 4) * ksteps) * 32 + lane) * 16;
  const _Float16* bp1 = bp0 + bstride;
  const _Float16* bp2 = bp1 + bstride;
  const _Float16* bp3 = bp2 + bstride;

  v8f acc0[4] = {};                             // M-tile 0
  v8f acc1[4] = {};                             // M-tile 1

  for (int ks = 0; ks < ksteps; ++ks) {
    // A fragments: 2 runs of 8 halves each, both M-tiles
    const v8h lo0 = *(const v8h*)(Ar0 + ks * 32);
    const v8h hh0 = *(const v8h*)(Ar0 + ks * 32 + 16);
    const v8h lo1 = *(const v8h*)(Ar1 + ks * 32);
    const v8h hh1 = *(const v8h*)(Ar1 + ks * 32 + 16);
    const v16h a0 = __builtin_shufflevector(lo0, hh0, 0, 1, 2, 3, 4, 5, 6, 7,
                                            8, 9, 10, 11, 12, 13, 14, 15);
    const v16h a1 = __builtin_shufflevector(lo1, hh1, 0, 1, 2, 3, 4, 5, 6, 7,
                                            8, 9, 10, 11, 12, 13, 14, 15);
    const v16h b0 = *(const v16h*)(bp0 + (size_t)ks * 512);
    const v16h b1v = *(const v16h*)(bp1 + (size_t)ks * 512);
    const v16h b2v = *(const v16h*)(bp2 + (size_t)ks * 512);
    const v16h b3v = *(const v16h*)(bp3 + (size_t)ks * 512);
    acc0[0] = __builtin_amdgcn_wmma_f32_16x16x32_f16(false, a0, false, b0,  (short)0, acc0[0], false, false);
    acc1[0] = __builtin_amdgcn_wmma_f32_16x16x32_f16(false, a1, false, b0,  (short)0, acc1[0], false, false);
    acc0[1] = __builtin_amdgcn_wmma_f32_16x16x32_f16(false, a0, false, b1v, (short)0, acc0[1], false, false);
    acc1[1] = __builtin_amdgcn_wmma_f32_16x16x32_f16(false, a1, false, b1v, (short)0, acc1[1], false, false);
    acc0[2] = __builtin_amdgcn_wmma_f32_16x16x32_f16(false, a0, false, b2v, (short)0, acc0[2], false, false);
    acc1[2] = __builtin_amdgcn_wmma_f32_16x16x32_f16(false, a1, false, b2v, (short)0, acc1[2], false, false);
    acc0[3] = __builtin_amdgcn_wmma_f32_16x16x32_f16(false, a0, false, b3v, (short)0, acc0[3], false, false);
    acc1[3] = __builtin_amdgcn_wmma_f32_16x16x32_f16(false, a1, false, b3v, (short)0, acc1[3], false, false);
  }

  // D layout: VGPR r, lanes 0-15: M=r, N=lane; lanes 16-31: M=8+r, N=lane-16
#pragma unroll
  for (int i = 0; i < 4; ++i) {
    const int n0 = (ng * 4 + i) * 16 + (lane & 15);
    const float bias = b1[n0] + (b2 ? b2[n0] : 0.0f);
#pragma unroll
    for (int r = 0; r < 8; ++r) {
      const int m0 = mt2 * 32 + r + hi * 8;
      float v0 = acc0[i][r] + bias;
      float v1 = acc1[i][r] + bias;
      if (relu) { v0 = fmaxf(v0, 0.0f); v1 = fmaxf(v1, 0.0f); }
      if (C32) {
        C32[(size_t)m0 * ldc + n0]        = v0;
        C32[(size_t)(m0 + 16) * ldc + n0] = v1;
      }
      if (C16) {
        C16[(size_t)m0 * ldc + n0]        = (_Float16)v0;
        C16[(size_t)(m0 + 16) * ldc + n0] = (_Float16)v1;
      }
    }
  }
}

// ---------------------------------------------------------------------------
// Edge-replicated delta features (window = 2). Reads f32 source channel band,
// writes f32 (for the next delta pass) and f16 (for the LSTM input GEMMs).
// ---------------------------------------------------------------------------
__global__ void delta_kernel(float* __restrict__ feat32, _Float16* __restrict__ feat16,
                             int src_off, int dst_off) {
  const int idx = blockIdx.x * blockDim.x + threadIdx.x;    // B*T*256
  if (idx >= BATCH * T_LEN * HID) return;
  const int c = idx & 255;
  const int t = (idx >> 8) & (T_LEN - 1);
  const int b = idx >> 18;
  const float* s = feat32 + (size_t)b * T_LEN * 768 + src_off + c;
  float z = 0.0f;
#pragma unroll
  for (int j = 1; j <= 2; ++j) {
    const int tp = (t + j > T_LEN - 1) ? T_LEN - 1 : t + j;
    const int tm = (t - j < 0) ? 0 : t - j;
    z += (s[(size_t)tp * 768] - s[(size_t)tm * 768]) * (0.5f / (float)j);
  }
  const size_t o = (size_t)b * T_LEN * 768 + (size_t)t * 768 + dst_off + c;
  feat32[o] = z;
  feat16[o] = (_Float16)z;
}

// ---------------------------------------------------------------------------
// Stable argsort(-lens) by rank counting (B=32, one wave).
// ---------------------------------------------------------------------------
__global__ void sort_kernel(const int* __restrict__ lens, int* __restrict__ order,
                            int* __restrict__ lens_sorted, float* __restrict__ order_out) {
  const int tid = threadIdx.x;
  if (tid >= BATCH) return;
  const int len = lens[tid];
  int rank = 0;
  for (int j = 0; j < BATCH; ++j) {
    const int lj = lens[j];
    rank += (lj > len) || (lj == len && j < tid);
  }
  order[rank]       = tid;
  lens_sorted[rank] = len;
  order_out[rank]   = (float)tid;
}

// ---------------------------------------------------------------------------
// Row maps: GEMM row m = t*32 + b (sorted batch b). Folds order permutation
// and per-sequence time reversal (backward dir) into the gather.
// ---------------------------------------------------------------------------
__global__ void rowmap_kernel(const int* __restrict__ order, const int* __restrict__ lens_sorted,
                              int* __restrict__ rmf, int* __restrict__ rmb) {
  const int idx = blockIdx.x * blockDim.x + threadIdx.x;    // T*B
  if (idx >= T_LEN * BATCH) return;
  const int b = idx & (BATCH - 1);
  const int t = idx >> 5;
  const int ob = order[b];
  rmf[idx] = ob * T_LEN + t;
  int tt = lens_sorted[b] - 1 - t;
  tt = tt < 0 ? 0 : (tt > T_LEN - 1 ? T_LEN - 1 : tt);
  rmb[idx] = ob * T_LEN + tt;
}

// ---------------------------------------------------------------------------
// Persistent LSTM recurrence. Batch rows are independent sequences, so split
// into 4 blocks: blockIdx.x = dir*2 + batch-tile. 512 threads = 16 waves;
// wave nh owns a 16(batch) x 16(hidden) block and computes all 4 gate tiles
// for it (i,f,g,o live at N = nh*16 + {0,256,512,768}) -> nonlinearity is
// wave-local. h in LDS (f16, 8 KB), c in registers, gx gate values
// prefetched into registers before the matmul so their latency hides behind
// the 32 WMMAs of the step.
// ---------------------------------------------------------------------------
__global__ void lstm_kernel(const float* __restrict__ gxf, const float* __restrict__ gxb,
                            const _Float16* __restrict__ WhhPf, const _Float16* __restrict__ WhhPb,
                            const float* __restrict__ h0, const float* __restrict__ c0,
                            const int* __restrict__ lens_sorted, float* __restrict__ out) {
  const int dir = blockIdx.x >> 1;
  const int mt  = blockIdx.x & 1;
  const float*    gx = dir ? gxb : gxf;
  const _Float16* Wp = dir ? WhhPb : WhhPf;

  __shared__ _Float16 h_buf[16][HID];      // 8 KB
  __shared__ int lens_sh[16];

  const int tid  = threadIdx.x;            // 0..511
  const int lane = tid & 31;
  const int nh   = tid >> 5;               // hidden tile 0..15
  const int hi   = lane >> 4;
  const int arow = lane & 15;              // local batch row for A fragment
  const int nloc = nh * 16 + (lane & 15);  // hidden column owned in epilogue

  if (tid < 16) lens_sh[tid] = lens_sorted[mt * 16 + tid];
#pragma unroll
  for (int i = 0; i < 8; ++i) {
    const int idx = tid * 8 + i;           // 16*256 = 4096 elements
    ((_Float16*)h_buf)[idx] = (_Float16)h0[(size_t)dir * BATCH * HID + (size_t)mt * 16 * HID + idx];
  }
  float creg[8];
#pragma unroll
  for (int r = 0; r < 8; ++r) {
    const int m = mt * 16 + r + hi * 8;
    creg[r] = c0[(size_t)dir * BATCH * HID + m * HID + nloc];
  }
  // B fragment base pointers, one per gate (n_tile = gate*16 + nh)
  const _Float16* bg0 = Wp + ((size_t)((0 * 16 + nh) * 8) * 32 + lane) * 16;
  const _Float16* bg1 = Wp + ((size_t)((1 * 16 + nh) * 8) * 32 + lane) * 16;
  const _Float16* bg2 = Wp + ((size_t)((2 * 16 + nh) * 8) * 32 + lane) * 16;
  const _Float16* bg3 = Wp + ((size_t)((3 * 16 + nh) * 8) * 32 + lane) * 16;
  const float* gxbase = gx + (size_t)(mt * 16 + hi * 8) * 1024 + nloc;
  __syncthreads();

  for (int t = 0; t < T_LEN; ++t) {
    // Prefetch this step's gate biases (independent of h): 32 loads issued
    // up front, consumed only after the WMMA loop.
    const float* gp = gxbase + (size_t)t * BATCH * 1024;
    float gv0[8], gv1[8], gv2[8], gv3[8];
#pragma unroll
    for (int r = 0; r < 8; ++r) {
      gv0[r] = gp[r * 1024 + 0];
      gv1[r] = gp[r * 1024 + 256];
      gv2[r] = gp[r * 1024 + 512];
      gv3[r] = gp[r * 1024 + 768];
    }

    v8f acc[4] = {};
#pragma unroll
    for (int ks = 0; ks < HID / 32; ++ks) {
      // A fragment: two contiguous 8-half runs -> 2x ds_load_b128
      const v8h lo = *(const v8h*)(&h_buf[arow][ks * 32 + hi * 8]);
      const v8h hh = *(const v8h*)(&h_buf[arow][ks * 32 + hi * 8 + 16]);
      const v16h a = __builtin_shufflevector(lo, hh, 0, 1, 2, 3, 4, 5, 6, 7,
                                             8, 9, 10, 11, 12, 13, 14, 15);
      const v16h b0 = *(const v16h*)(bg0 + ks * 512);
      const v16h b1 = *(const v16h*)(bg1 + ks * 512);
      const v16h b2 = *(const v16h*)(bg2 + ks * 512);
      const v16h b3 = *(const v16h*)(bg3 + ks * 512);
      acc[0] = __builtin_amdgcn_wmma_f32_16x16x32_f16(false, a, false, b0, (short)0, acc[0], false, false);
      acc[1] = __builtin_amdgcn_wmma_f32_16x16x32_f16(false, a, false, b1, (short)0, acc[1], false, false);
      acc[2] = __builtin_amdgcn_wmma_f32_16x16x32_f16(false, a, false, b2, (short)0, acc[2], false, false);
      acc[3] = __builtin_amdgcn_wmma_f32_16x16x32_f16(false, a, false, b3, (short)0, acc[3], false, false);
    }
    __syncthreads();                       // all h_buf reads for step t done

#pragma unroll
    for (int r = 0; r < 8; ++r) {
      const float ig = sigmoidf_(acc[0][r] + gv0[r]);
      const float fg = sigmoidf_(acc[1][r] + gv1[r]);
      const float gg = tanhf    (acc[2][r] + gv2[r]);
      const float og = sigmoidf_(acc[3][r] + gv3[r]);
      const float c  = fg * creg[r] + ig * gg;
      creg[r] = c;
      const float h = og * tanhf(c);
      const int lm = r + hi * 8;           // local batch row
      h_buf[lm][nloc] = (_Float16)h;       // disjoint 16x16 block per wave
      const int m = mt * 16 + lm;
      if (dir == 0) {
        out[((size_t)m * T_LEN + t) * 512 + nloc] = h;
      } else {
        const int L = lens_sh[lm];
        if (t < L) out[((size_t)m * T_LEN + (L - 1 - t)) * 512 + 256 + nloc] = h;
      }
    }
    __syncthreads();                       // new h visible before next step
  }
}

// ---------------------------------------------------------------------------
// Zero padded timesteps (pad_packed_sequence semantics); also covers the
// backward half that was never stored for t >= len.
// ---------------------------------------------------------------------------
__global__ void finalize_kernel(float* __restrict__ out, const int* __restrict__ lens_sorted) {
  const size_t idx = (size_t)blockIdx.x * blockDim.x + threadIdx.x;
  if (idx >= (size_t)BATCH * T_LEN * 512) return;
  const int t = (int)((idx >> 9) & (T_LEN - 1));
  const int b = (int)(idx >> 19);
  if (t >= lens_sorted[b]) out[idx] = 0.0f;
}

// ---------------------------------------------------------------------------
extern "C" void kernel_launch(void* const* d_in, const int* in_sizes, int n_in,
                              void* d_out, int out_size, void* d_ws, size_t ws_size,
                              hipStream_t stream) {
  (void)in_sizes; (void)n_in; (void)out_size; (void)ws_size;
  const float* x     = (const float*)d_in[0];
  const int*   xlen  = (const int*)  d_in[1];
  const float* encW0 = (const float*)d_in[2];
  const float* encB0 = (const float*)d_in[3];
  const float* encW1 = (const float*)d_in[4];
  const float* encB1 = (const float*)d_in[5];
  const float* encW2 = (const float*)d_in[6];
  const float* encB2 = (const float*)d_in[7];
  const float* WihF  = (const float*)d_in[8];
  const float* WhhF  = (const float*)d_in[9];
  const float* bihF  = (const float*)d_in[10];
  const float* bhhF  = (const float*)d_in[11];
  const float* WihB  = (const float*)d_in[12];
  const float* WhhB  = (const float*)d_in[13];
  const float* bihB  = (const float*)d_in[14];
  const float* bhhB  = (const float*)d_in[15];
  const float* h0    = (const float*)d_in[16];
  const float* c0    = (const float*)d_in[17];
  float* out = (float*)d_out;

  char* wsp = (char*)d_ws;
  auto alloc = [&](size_t bytes) -> char* {
    char* p = wsp; wsp += (bytes + 255) & ~(size_t)255; return p;
  };
  const size_t MROWS = (size_t)BATCH * T_LEN;              // 32768
  float*    feat32 = (float*)   alloc(MROWS * 768 * 4);    // f32 h/d1/d2 (delta chain)
  _Float16* feat16 = (_Float16*)alloc(MROWS * 768 * 2);    // f16 copy (GEMM A operand)
  float*    gxf    = (float*)   alloc(MROWS * 1024 * 4);   // [t*32+b, 4H]
  float*    gxb    = (float*)   alloc(MROWS * 1024 * 4);
  _Float16* xpad   = (_Float16*)alloc(MROWS * 96 * 2);     // zero-padded f16 x
  _Float16* tmp0   = (_Float16*)gxf;                       // enc f16 scratch (reused before gx)
  _Float16* tmp1   = (_Float16*)gxb;
  _Float16* pW0   = (_Float16*)alloc((512 / 16) * 3  * 1024);   // K=80 -> 3 ksteps
  _Float16* pW1   = (_Float16*)alloc((512 / 16) * 16 * 1024);
  _Float16* pW2   = (_Float16*)alloc((256 / 16) * 16 * 1024);
  _Float16* pWihF = (_Float16*)alloc((1024 / 16) * 24 * 1024);
  _Float16* pWihB = (_Float16*)alloc((1024 / 16) * 24 * 1024);
  _Float16* pWhhF = (_Float16*)alloc((1024 / 16) * 8  * 1024);
  _Float16* pWhhB = (_Float16*)alloc((1024 / 16) * 8  * 1024);
  int* order  = (int*)alloc(BATCH * 4);
  int* lens_s = (int*)alloc(BATCH * 4);
  int* rmf    = (int*)alloc(MROWS * 4);
  int* rmb    = (int*)alloc(MROWS * 4);

  auto pack = [&](const float* W, _Float16* P, int N, int K) {
    const int total = (N >> 4) * ((K + 31) >> 5) * 32;
    pack_w_kernel<<<(total + 255) / 256, 256, 0, stream>>>(W, P, N, K);
  };
  pack(encW0, pW0, 512, 80);
  pack(encW1, pW1, 512, 512);
  pack(encW2, pW2, 256, 512);
  pack(WihF, pWihF, 1024, 768);
  pack(WihB, pWihB, 1024, 768);
  pack(WhhF, pWhhF, 1024, 256);
  pack(WhhB, pWhhB, 1024, 256);

  pad_x_kernel<<<(int)((MROWS * 96 + 255) / 256), 256, 0, stream>>>(x, xpad);
  sort_kernel<<<1, 32, 0, stream>>>(xlen, order, lens_s, out + (size_t)BATCH * T_LEN * 512);
  rowmap_kernel<<<(int)((MROWS + 255) / 256), 256, 0, stream>>>(order, lens_s, rmf, rmb);

  // Encoder MLP (relu after each layer). Grids sized exactly: (M/32)*(N/64)/8.
  // Intermediates are f16 only; layer 2 writes h as f32 (delta chain) + f16.
  wmma_gemm_kernel<<<1024, 256, 0, stream>>>(xpad, nullptr, pW0, encB0, nullptr,
                                             nullptr, tmp0, 32768, 512, 96, 96, 512, 1);
  wmma_gemm_kernel<<<1024, 256, 0, stream>>>(tmp0, nullptr, pW1, encB1, nullptr,
                                             nullptr, tmp1, 32768, 512, 512, 512, 512, 1);
  wmma_gemm_kernel<<<512, 256, 0, stream>>>(tmp1, nullptr, pW2, encB2, nullptr,
                                            feat32, feat16, 32768, 256, 512, 512, 768, 1);
  // Delta features into columns [256:512] and [512:768] (f32 + f16 copies)
  delta_kernel<<<32768, 256, 0, stream>>>(feat32, feat16, 0,   256);
  delta_kernel<<<32768, 256, 0, stream>>>(feat32, feat16, 256, 512);

  // Hoisted input projections (gathered by rowmap; bias = b_ih + b_hh folded in)
  wmma_gemm_kernel<<<2048, 256, 0, stream>>>(feat16, rmf, pWihF, bihF, bhhF,
                                             gxf, nullptr, 32768, 1024, 768, 768, 1024, 0);
  wmma_gemm_kernel<<<2048, 256, 0, stream>>>(feat16, rmb, pWihB, bihB, bhhB,
                                             gxb, nullptr, 32768, 1024, 768, 768, 1024, 0);

  // Sequential recurrence: 4 persistent workgroups (2 dirs x 2 batch tiles).
  lstm_kernel<<<4, 512, 0, stream>>>(gxf, gxb, pWhhF, pWhhB, h0, c0, lens_s, out);

  // Mask padded steps to zero.
  finalize_kernel<<<65536, 256, 0, stream>>>(out, lens_s);
}
